// LayerNormLSTMCell_47579647705748
// MI455X (gfx1250) — matrix operand order repaired and checked by reading, not compiled, not run on hardware
//
#include <hip/hip_runtime.h>
#include <hip/hip_bf16.h>

// Problem constants (from reference setup_inputs)
#define B_    128
#define T_    512
#define D_    512
#define U_    512
#define NG    2048          // 4 * U  (stacked gate outputs)
#define KT    1024          // U + D  (cat dimension)
#define EPSV  1e-3f

// ---- WMMA fragment types (CDNA5 wave32, v_wmma_f32_16x16x32_bf16) ----
typedef __attribute__((ext_vector_type(16))) __bf16 v16bf;
typedef __attribute__((ext_vector_type(8)))  float  v8f;

union Frag { v16bf v; uint4 q[2]; };

__device__ __forceinline__ unsigned short f2bf(float f) {
    unsigned int u = __float_as_uint(f);
    u += 0x7FFFu + ((u >> 16) & 1u);      // round-to-nearest-even
    return (unsigned short)(u >> 16);
}
__device__ __forceinline__ float sigmoidf_(float x) { return 1.0f / (1.0f + __expf(-x)); }

// ---------------------------------------------------------------------------
// Prep kernel: repack stacked gate weights [4][U][U+D] fp32 into bf16
// "fragment-native" B layout: wfrag[(((kb*128 + nt)*32 + lane)*16) + i]
//   K = kb*32 + (lane>>4)*16 + i ,  N = nt*16 + (lane&15)  (contiguous k order, same as A)
// ---------------------------------------------------------------------------
__global__ void prep_wfrag(const float* __restrict__ Wf, const float* __restrict__ Wg,
                           const float* __restrict__ Wi, const float* __restrict__ Wo,
                           unsigned short* __restrict__ wfrag) {
    int idx  = blockIdx.x * blockDim.x + threadIdx.x;   // exactly 2^21 threads
    int i    = idx & 15;
    int lane = (idx >> 4) & 31;
    int nt   = (idx >> 9) & 127;
    int kb   = idx >> 16;
    int K    = kb * 32 + ((lane >> 4) << 4) + i;
    int n    = nt * 16 + (lane & 15);
    int g    = n >> 9;
    int u    = n & 511;
    const float* W = (g == 0) ? Wf : (g == 1) ? Wg : (g == 2) ? Wi : Wo;
    wfrag[idx] = f2bf(W[(size_t)u * KT + K]);
}

__global__ void prep_bias(const float* __restrict__ bf, const float* __restrict__ bg,
                          const float* __restrict__ bi, const float* __restrict__ bo,
                          float* __restrict__ biasAll) {
    int n = blockIdx.x * blockDim.x + threadIdx.x;
    if (n < NG) {
        int g = n >> 9, u = n & 511;
        const float* b = (g == 0) ? bf : (g == 1) ? bg : (g == 2) ? bi : bo;
        biasAll[n] = b[u];
    }
}

// ---------------------------------------------------------------------------
// Recurrent kernel: one workgroup per 16-row batch tile, full time loop inside.
// 1024 threads = 32 waves; each wave owns 4 N-tiles (16 cols each) of NG=2048.
// ---------------------------------------------------------------------------
#define CSTR 1032   // ushort row stride of cat (1024 + 8): 16B-aligned rows, bank shift 4
#define ZSTR 2052   // float row stride of z   (2048 + 4)
#define CBST 516    // float row stride of c   (512 + 4)
#define CAT_BYTES (16 * CSTR * 2)                 // 33024
#define Z_BYTES   (16 * ZSTR * 4)                 // 131328
#define C_BYTES   (16 * CBST * 4)                 // 33024
#define LDS_BYTES (CAT_BYTES + Z_BYTES + C_BYTES) // 197376

__global__ void __launch_bounds__(1024)
lnlstm_kernel(const float* __restrict__ x, const float* __restrict__ h0,
              const float* __restrict__ c0,
              const unsigned short* __restrict__ wfrag,
              const float* __restrict__ biasAll,
              const float* __restrict__ gamma, const float* __restrict__ beta,
              float* __restrict__ out) {
    extern __shared__ char smem[];
    unsigned short* catA = (unsigned short*)smem;                 // bf16 [16][CSTR]  (cat = [h | x_t])
    float*          zbuf = (float*)(smem + CAT_BYTES);            // fp32 [16][ZSTR]  (gate pre/post-act)
    float*          cbuf = (float*)(smem + CAT_BYTES + Z_BYTES);  // fp32 [16][CBST]  (carried c, LN'd)

    const int tid   = threadIdx.x;
    const int lane  = tid & 31;
    const int wave  = tid >> 5;
    const int half  = lane >> 4;    // 0: lanes 0-15, 1: lanes 16-31
    const int l16   = lane & 15;
    const int bbase = blockIdx.x * 16;

    // ---- init carried state ----
    for (int idx = tid; idx < 16 * U_; idx += 1024) {
        int row = idx >> 9, u = idx & 511;
        catA[row * CSTR + u] = f2bf(h0[(size_t)(bbase + row) * U_ + u]);
        cbuf[row * CBST + u] = c0[(size_t)(bbase + row) * U_ + u];
    }
    __syncthreads();

    for (int t = 0; t < T_; ++t) {
        // ---- phase 1: x_t -> bf16 into cat[:, U:U+D] ----
        for (int idx = tid; idx < 16 * D_; idx += 1024) {
            int row = idx >> 9, u = idx & 511;
            float xv = x[((size_t)(bbase + row) * T_ + t) * D_ + u];
            catA[row * CSTR + U_ + u] = f2bf(xv);
        }
        __syncthreads();

        // ---- phase 2: z = cat @ W^T via WMMA (K = 1024, 4 tiles per wave) ----
        v8f acc0 = {}, acc1 = {}, acc2 = {}, acc3 = {};
        {
            const int arow  = l16;
            const int abase = half * 16;
            for (int kb = 0; kb < 32; ++kb) {
                Frag a;  // A fragment: contiguous k = 16*half + s, two 16B LDS reads
                a.q[0] = *(const uint4*)(catA + arow * CSTR + kb * 32 + abase);
                a.q[1] = *(const uint4*)(catA + arow * CSTR + kb * 32 + 8 + abase);
                const unsigned short* wp =
                    wfrag + ((size_t)(kb * 128 + wave * 4) * 32 + lane) * 16;
#define GEMM_TILE(J, ACC)                                                        \
                do {                                                             \
                    Frag b;                                                      \
                    const uint4* pb = (const uint4*)(wp + (J) * 512);            \
                    b.q[0] = pb[0]; b.q[1] = pb[1];                              \
                    ACC = __builtin_amdgcn_wmma_f32_16x16x32_bf16(               \
                        false, a.v, false, b.v, (short)0, ACC, false, false);    \
                } while (0)
                GEMM_TILE(0, acc0);
                GEMM_TILE(1, acc1);
                GEMM_TILE(2, acc2);
                GEMM_TILE(3, acc3);
#undef GEMM_TILE
            }
        }
        // spill accumulators (+bias) to LDS: C layout rows = r + half*8, col = nt*16 + l16
#define STORE_TILE(J, ACC)                                                       \
        do {                                                                     \
            int n = (wave * 4 + (J)) * 16 + l16;                                 \
            float bs = biasAll[n];                                               \
            _Pragma("unroll")                                                    \
            for (int r = 0; r < 8; ++r)                                          \
                zbuf[(r + half * 8) * ZSTR + n] = ACC[r] + bs;                   \
        } while (0)
        STORE_TILE(0, acc0);
        STORE_TILE(1, acc1);
        STORE_TILE(2, acc2);
        STORE_TILE(3, acc3);
#undef STORE_TILE
        __syncthreads();

        // ---- phase 3: per-gate LayerNorm + activation (64 groups of 16 lanes) ----
        {
            int gid = tid >> 4;          // 0..63
            int row = gid >> 2;
            int g   = gid & 3;
            float* zr = zbuf + row * ZSTR + g * U_;
            float s = 0.f, s2 = 0.f;
            for (int j = 0; j < 32; ++j) {
                float zv = zr[j * 16 + l16];
                s += zv; s2 += zv * zv;
            }
            for (int m = 1; m < 16; m <<= 1) {
                s  += __shfl_xor(s,  m, 16);
                s2 += __shfl_xor(s2, m, 16);
            }
            float mean = s * (1.0f / U_);
            float var  = s2 * (1.0f / U_) - mean * mean;
            float rstd = rsqrtf(var + EPSV);
            for (int j = 0; j < 32; ++j) {
                int u = j * 16 + l16;
                float zn = (zr[u] - mean) * rstd * gamma[u] + beta[u];
                zr[u] = (g == 1) ? tanhf(zn) : sigmoidf_(zn);   // gate 1 = g (tanh)
            }
        }
        __syncthreads();

        // ---- phase 4: c update, c-LayerNorm, h output (one wave per row) ----
        if (wave < 16) {
            int row = wave;
            float* zr = zbuf + row * ZSTR;
            float* cr = cbuf + row * CBST;
            float cv[16];
            float s = 0.f, s2 = 0.f;
            for (int j = 0; j < 16; ++j) {
                int u = lane + j * 32;
                float fg = zr[u];                 // forget (sigmoid)
                float gg = zr[U_ + u];            // cell   (tanh)
                float ig = zr[2 * U_ + u];        // input  (sigmoid)
                float cn = cr[u] * fg + gg * ig;
                cv[j] = cn; s += cn; s2 += cn * cn;
            }
            for (int m = 1; m < 32; m <<= 1) {
                s  += __shfl_xor(s,  m, 32);
                s2 += __shfl_xor(s2, m, 32);
            }
            float mean = s * (1.0f / U_);
            float var  = s2 * (1.0f / U_) - mean * mean;
            float rstd = rsqrtf(var + EPSV);
            for (int j = 0; j < 16; ++j) {
                int u = lane + j * 32;
                float cln = (cv[j] - mean) * rstd * gamma[u] + beta[u];
                cr[u] = cln;                               // carried c is the normalized one
                float o  = zr[3 * U_ + u];                 // output gate (sigmoid)
                float hv = o * tanhf(cln);
                out[((size_t)(bbase + row) * T_ + t) * U_ + u] = hv;
                catA[row * CSTR + u] = f2bf(hv);           // h for next step's A matrix
            }
        }
        __syncthreads();
    }
}

// ---------------------------------------------------------------------------
extern "C" void kernel_launch(void* const* d_in, const int* in_sizes, int n_in,
                              void* d_out, int out_size, void* d_ws, size_t ws_size,
                              hipStream_t stream) {
    (void)in_sizes; (void)n_in; (void)out_size; (void)ws_size;
    const float* x     = (const float*)d_in[0];
    const float* h0    = (const float*)d_in[1];
    const float* c0    = (const float*)d_in[2];
    const float* Wf    = (const float*)d_in[3];
    const float* bf    = (const float*)d_in[4];
    const float* Wg    = (const float*)d_in[5];
    const float* bg    = (const float*)d_in[6];
    const float* Wi    = (const float*)d_in[7];
    const float* bi    = (const float*)d_in[8];
    const float* Wo    = (const float*)d_in[9];
    const float* bo    = (const float*)d_in[10];
    const float* gamma = (const float*)d_in[11];
    const float* beta  = (const float*)d_in[12];

    unsigned short* wfrag   = (unsigned short*)d_ws;                       // 4 MB bf16 weights
    float*          biasAll = (float*)((char*)d_ws + (size_t)KT * NG * 2); // 8 KB stacked bias

    prep_wfrag<<<2048, 1024, 0, stream>>>(Wf, Wg, Wi, Wo, wfrag);
    prep_bias<<<2, 1024, 0, stream>>>(bf, bg, bi, bo, biasAll);

    lnlstm_kernel<<<B_ / 16, 1024, LDS_BYTES, stream>>>(
        x, h0, c0, wfrag, biasAll, gamma, beta, (float*)d_out);
}
